// SCDecoderHY_22686017257978
// MI455X (gfx1250) — matrix operand order repaired and checked by reading, not compiled, not run on hardware
//
#include <hip/hip_runtime.h>
#include <cstddef>

#define Bsz  32
#define Nseq 512
#define Dm   128
#define Hm   256

typedef __attribute__((ext_vector_type(16))) __bf16 v16bf;
typedef __attribute__((ext_vector_type(8)))  float  v8f;

struct B32x2 { uint4 a, b; };

__device__ __forceinline__ __bf16 f2bf(float f) {
  unsigned u = __builtin_bit_cast(unsigned, f);
  u += 0x7FFFu + ((u >> 16) & 1u);                 // round-to-nearest-even
  unsigned short s = (unsigned short)(u >> 16);
  return __builtin_bit_cast(__bf16, s);
}
__device__ __forceinline__ float bf2f(__bf16 x) {
  unsigned short s = __builtin_bit_cast(unsigned short, x);
  unsigned u = (unsigned)s << 16;
  return __builtin_bit_cast(float, u);
}
__device__ __forceinline__ v16bf mk_v16bf(uint4 q0, uint4 q1) {
  B32x2 t{q0, q1};
  return __builtin_bit_cast(v16bf, t);
}

// ---------------------------------------------------------------------------
// Init: level-0 embeddings, stored bf16.
// ---------------------------------------------------------------------------
__global__ void init_emb_kernel(const int* __restrict__ y,
                                const float* __restrict__ emb_enc,
                                const float* __restrict__ emb_dec,
                                __bf16* __restrict__ eco0,
                                __bf16* __restrict__ ech0) {
  int t = blockIdx.x * blockDim.x + threadIdx.x;          // over B*N*(D/8)
  const int total = Bsz * Nseq * (Dm / 8);
  if (t >= total) return;
  int dv   = t & 15;                                      // Dm/8 == 16
  int rowi = t >> 4;                                      // b*N + n
  int k0   = dv * 8;
  const float* se = emb_enc + 2 * Dm + k0;
  const float* sd = emb_dec + (size_t)y[rowi] * Dm + k0;
  __bf16 eo[8], ed[8];
  #pragma unroll
  for (int e = 0; e < 8; ++e) { eo[e] = f2bf(se[e]); ed[e] = f2bf(sd[e]); }
  *(uint4*)(eco0 + (size_t)rowi * Dm + k0) = *(const uint4*)eo;
  *(uint4*)(ech0 + (size_t)rowi * Dm + k0) = *(const uint4*)ed;
}

__global__ void pack_labels_kernel(const float* __restrict__ lab,
                                   __bf16* __restrict__ labbf) {
  int t = threadIdx.x;                                    // 256 = 2*Dm
  labbf[t] = f2bf(lab[t]);
}

// ---------------------------------------------------------------------------
// Pack (K x Ncols) fp32 weight into bf16 WMMA B-fragment order:
//  packed[((kt*numNt+nt)*32+lane)*16+e] = W[kt*32+(lane/16)*8+(e%8)+(e/8)*16][nt*16+lane%16]
// ---------------------------------------------------------------------------
__global__ void pack_w_kernel(const float* __restrict__ W,
                              __bf16* __restrict__ Wp, int K, int Ncols) {
  int t = blockIdx.x * blockDim.x + threadIdx.x;
  int total = K * Ncols;
  if (t >= total) return;
  int e    = t & 15;
  int lane = (t >> 4) & 31;
  int tile = t >> 9;
  int numNt = Ncols >> 4;
  int kt = tile / numNt, nt = tile - kt * numNt;
  int k   = kt * 32 + (lane >> 4) * 8 + (e & 7) + (e >> 3) * 16;
  int col = nt * 16 + (lane & 15);
  Wp[t] = f2bf(W[(size_t)k * Ncols + col]);
}

// ---------------------------------------------------------------------------
// Fused MLP:  out = relu(X @ W1 + b1) @ W2 + b2   (all bf16 in, bf16 out)
//   X rows r = b*h + i : virtual concat [e[b,2i], e[b,2i+1], (u_emb)]
//   Block = 32 rows (2 row-tiles), 128 threads (4 waves). Each B (weight)
//   fragment is reused for 2 WMMAs (both row-tiles) to halve L2 weight
//   traffic and double WMMA density between waits. X + hidden tiles in LDS.
// ---------------------------------------------------------------------------
template <int K1>
__global__ __launch_bounds__(128)
void mlp_wmma_kernel(const __bf16* __restrict__ eparent,  // (B, n, D) bf16
                     const int*    __restrict__ u1hp,     // (B, h) or null
                     const __bf16* __restrict__ elab,     // (2, D) bf16
                     const __bf16* __restrict__ W1p, const float* __restrict__ b1,
                     const __bf16* __restrict__ W2p, const float* __restrict__ b2,
                     __bf16* __restrict__ echild,         // (B, h, D) bf16
                     int n, int h) {
  __shared__ __bf16 xtile[32 * K1];                       // <= 24 KB
  __shared__ __bf16 hid[32 * Hm];                         // 16 KB
  const int tid  = threadIdx.x;
  const int lane = tid & 31;
  const int wave = tid >> 5;
  const int rbase = blockIdx.x * 32;

  // ---- phase 0: stage concatenated X tile into LDS (16-byte chunks)
  constexpr int CPR = K1 / 8;                             // chunks per row
  #pragma unroll
  for (int c0 = 0; c0 < 32 * CPR; c0 += 128) {
    int c  = c0 + tid;
    int m  = c / CPR;
    int k0 = (c - m * CPR) * 8;
    int r  = rbase + m;
    int b  = r / h, i = r - b * h;
    const __bf16* src;
    if (k0 < 128)
      src = eparent + (size_t)(b * n + 2 * i) * Dm + k0;
    else if (k0 < 256)
      src = eparent + (size_t)(b * n + 2 * i + 1) * Dm + (k0 - 128);
    else
      src = elab + (size_t)u1hp[b * h + i] * Dm + (k0 - 256);
    *(uint4*)(&xtile[m * K1 + k0]) = *(const uint4*)src;
  }
  __syncthreads();

  // ---- GEMM1: X(32xK1) @ W1(K1x256); wave w owns hidden cols [w*64, w*64+64)
  const int arow = lane & 15;
  const int aoff = (lane >> 4) * 8;
  v8f acc[2][4];
  #pragma unroll
  for (int rt = 0; rt < 2; ++rt)
    #pragma unroll
    for (int j = 0; j < 4; ++j)
      #pragma unroll
      for (int e = 0; e < 8; ++e) acc[rt][j][e] = 0.f;

  #pragma unroll
  for (int kt = 0; kt < K1 / 32; ++kt) {
    v16bf a[2];
    #pragma unroll
    for (int rt = 0; rt < 2; ++rt) {
      const __bf16* ap = &xtile[(rt * 16 + arow) * K1 + aoff + kt * 32];
      a[rt] = mk_v16bf(*(const uint4*)ap, *(const uint4*)(ap + 16));
    }
    #pragma unroll
    for (int j = 0; j < 4; ++j) {
      int nt = wave * 4 + j;
      const __bf16* wp = W1p + (((size_t)kt * (Hm / 16) + nt) * 32 + lane) * 16;
      v16bf bfr = mk_v16bf(((const uint4*)wp)[0], ((const uint4*)wp)[1]);
      #pragma unroll
      for (int rt = 0; rt < 2; ++rt)
        acc[rt][j] = __builtin_amdgcn_wmma_f32_16x16x32_bf16(
            false, a[rt], false, bfr, (short)0, acc[rt][j], false, false);
    }
  }

  // bias + ReLU -> hidden LDS (bf16)
  {
    int Mloc = (lane >> 4) * 8;
    #pragma unroll
    for (int j = 0; j < 4; ++j) {
      int col = (wave * 4 + j) * 16 + (lane & 15);
      float bb = b1[col];
      #pragma unroll
      for (int rt = 0; rt < 2; ++rt)
        #pragma unroll
        for (int rr = 0; rr < 8; ++rr) {
          float v = acc[rt][j][rr] + bb;
          hid[(rt * 16 + Mloc + rr) * Hm + col] = f2bf(v > 0.f ? v : 0.f);
        }
    }
  }
  __syncthreads();

  // ---- GEMM2: hidden(32x256) @ W2(256x128); wave w owns out cols [w*32,+32)
  v8f acc2[2][2];
  #pragma unroll
  for (int rt = 0; rt < 2; ++rt)
    #pragma unroll
    for (int j = 0; j < 2; ++j)
      #pragma unroll
      for (int e = 0; e < 8; ++e) acc2[rt][j][e] = 0.f;

  #pragma unroll
  for (int kt = 0; kt < Hm / 32; ++kt) {
    v16bf a[2];
    #pragma unroll
    for (int rt = 0; rt < 2; ++rt) {
      const __bf16* ap = &hid[(rt * 16 + arow) * Hm + aoff + kt * 32];
      a[rt] = mk_v16bf(*(const uint4*)ap, *(const uint4*)(ap + 16));
    }
    #pragma unroll
    for (int j = 0; j < 2; ++j) {
      int nt = wave * 2 + j;
      const __bf16* wp = W2p + (((size_t)kt * (Dm / 16) + nt) * 32 + lane) * 16;
      v16bf bfr = mk_v16bf(((const uint4*)wp)[0], ((const uint4*)wp)[1]);
      #pragma unroll
      for (int rt = 0; rt < 2; ++rt)
        acc2[rt][j] = __builtin_amdgcn_wmma_f32_16x16x32_bf16(
            false, a[rt], false, bfr, (short)0, acc2[rt][j], false, false);
    }
  }

  // bias -> bf16 -> global
  {
    int Mloc = (lane >> 4) * 8;
    #pragma unroll
    for (int j = 0; j < 2; ++j) {
      int col = (wave * 2 + j) * 16 + (lane & 15);
      float bb = b2[col];
      #pragma unroll
      for (int rt = 0; rt < 2; ++rt)
        #pragma unroll
        for (int rr = 0; rr < 8; ++rr) {
          int orow = rbase + rt * 16 + Mloc + rr;
          echild[(size_t)orow * Dm + col] = f2bf(acc2[rt][j][rr] + bb);
        }
    }
  }
}

// ---------------------------------------------------------------------------
// Leaf (n==1): logits, softmax, hard decisions, frozen-bit select.
// ---------------------------------------------------------------------------
__global__ void leaf_kernel(const __bf16* __restrict__ eco,
                            const __bf16* __restrict__ ech,
                            const float* __restrict__ wle, const float* __restrict__ ble,
                            const float* __restrict__ wld, const float* __restrict__ bld,
                            const int* __restrict__ f, const float* __restrict__ r,
                            int j, int* __restrict__ hpOut, float* __restrict__ out) {
  int b = threadIdx.x;
  if (b >= Bsz) return;
  const __bf16* xo = eco + (size_t)b * Dm;
  const __bf16* xc = ech + (size_t)b * Dm;
  float lo0 = ble[0], lo1 = ble[1], lc0 = bld[0], lc1 = bld[1];
  for (int k = 0; k < Dm; ++k) {
    float a = bf2f(xo[k]), c = bf2f(xc[k]);
    lo0 += a * wle[k * 2 + 0]; lo1 += a * wle[k * 2 + 1];
    lc0 += c * wld[k * 2 + 0]; lc1 += c * wld[k * 2 + 1];
  }
  float m  = fmaxf(lo0, lo1);
  float e0 = __expf(lo0 - m), e1 = __expf(lo1 - m);
  float pu0 = e0 / (e0 + e1);
  float mc  = fmaxf(lc0, lc1);
  float c0  = __expf(lc0 - mc), c1 = __expf(lc1 - mc);
  float s   = c0 + c1;
  float py0 = c0 / s, py1 = c1 / s;
  int hd_u  = (r[b * Nseq + j] > pu0) ? 1 : 0;
  int hd_uy = (py1 > py0) ? 1 : 0;
  int u = (f[b * Nseq + j] == 2) ? hd_uy : hd_u;
  hpOut[b] = u;
  out[b * Nseq + j] = (float)u;
  out[Bsz * Nseq + (size_t)(b * Nseq + j) * 2 + 0] = py0;
  out[Bsz * Nseq + (size_t)(b * Nseq + j) * 2 + 1] = py1;
}

// hp_parent[b,2k] = (l+r)%2 ; hp_parent[b,2k+1] = r
__global__ void combine_kernel(const int* __restrict__ hl, const int* __restrict__ hr,
                               int* __restrict__ dst, int h) {
  int t = blockIdx.x * blockDim.x + threadIdx.x;
  if (t >= Bsz * h) return;
  int b = t / h, k = t - b * h;
  int l = hl[t], rr = hr[t];
  dst[b * 2 * h + 2 * k]     = (l + rr) & 1;
  dst[b * 2 * h + 2 * k + 1] = rr;
}

// ---------------------------------------------------------------------------
// Host-side DFS (graph-capture time): enqueues the dependent kernel chain.
// ---------------------------------------------------------------------------
struct DecCtx {
  hipStream_t stream;
  const int *f; const float *r;
  const __bf16 *elab;
  const float *w_llr_enc, *b_llr_enc, *w_llr_dec, *b_llr_dec;
  const float *ce_b1, *ce_b2, *cd_b1, *cd_b2, *be_b1, *be_b2, *bd_b1, *bd_b2;
  const __bf16 *ceW1p, *ceW2p, *cdW1p, *cdW2p, *beW1p, *beW2p, *bdW1p, *bdW2p;
  __bf16* eco[10]; __bf16* ech[10];
  int*    hpL[10]; int*    hpR[10];
  float* out;
};

static void mlp_launch(DecCtx& c, const __bf16* eparent, const int* u1hp,
                       const __bf16* W1p, const float* b1,
                       const __bf16* W2p, const float* b2,
                       __bf16* echild, int n, int h, int K1) {
  int blocks = (Bsz * h) / 32;                            // h>=1 -> >=1 block
  if (K1 == 2 * Dm)
    mlp_wmma_kernel<2 * Dm><<<blocks, 128, 0, c.stream>>>(
        eparent, u1hp, c.elab, W1p, b1, W2p, b2, echild, n, h);
  else
    mlp_wmma_kernel<3 * Dm><<<blocks, 128, 0, c.stream>>>(
        eparent, u1hp, c.elab, W1p, b1, W2p, b2, echild, n, h);
}

static void decode_rec(DecCtx& c, int depth, int off, int n, int* hpOut) {
  if (n == 1) {
    leaf_kernel<<<1, 32, 0, c.stream>>>(c.eco[depth], c.ech[depth],
                                        c.w_llr_enc, c.b_llr_enc,
                                        c.w_llr_dec, c.b_llr_dec,
                                        c.f, c.r, off, hpOut, c.out);
    return;
  }
  int h = n / 2;
  // f-step
  mlp_launch(c, c.eco[depth], nullptr, c.ceW1p, c.ce_b1, c.ceW2p, c.ce_b2,
             c.eco[depth + 1], n, h, 2 * Dm);
  mlp_launch(c, c.ech[depth], nullptr, c.cdW1p, c.cd_b1, c.cdW2p, c.cd_b2,
             c.ech[depth + 1], n, h, 2 * Dm);
  decode_rec(c, depth + 1, off, h, c.hpL[depth + 1]);
  // g-step (conditioned on left child's hard bits)
  mlp_launch(c, c.eco[depth], c.hpL[depth + 1], c.beW1p, c.be_b1, c.beW2p, c.be_b2,
             c.eco[depth + 1], n, h, 3 * Dm);
  mlp_launch(c, c.ech[depth], c.hpL[depth + 1], c.bdW1p, c.bd_b1, c.bdW2p, c.bd_b2,
             c.ech[depth + 1], n, h, 3 * Dm);
  decode_rec(c, depth + 1, off + h, h, c.hpR[depth + 1]);
  combine_kernel<<<(Bsz * h + 127) / 128, 128, 0, c.stream>>>(
      c.hpL[depth + 1], c.hpR[depth + 1], hpOut, h);
}

extern "C" void kernel_launch(void* const* d_in, const int* in_sizes, int n_in,
                              void* d_out, int out_size, void* d_ws, size_t ws_size,
                              hipStream_t stream) {
  const int*   y           = (const int*)d_in[0];
  const int*   f           = (const int*)d_in[1];
  const float* r           = (const float*)d_in[2];
  const float* emb_obs_enc = (const float*)d_in[3];
  const float* emb_obs_dec = (const float*)d_in[4];
  const float* emb_labels  = (const float*)d_in[5];
  const float* w_llr_enc   = (const float*)d_in[6];
  const float* b_llr_enc   = (const float*)d_in[7];
  const float* w_llr_dec   = (const float*)d_in[8];
  const float* b_llr_dec   = (const float*)d_in[9];
  const float* ce_w1 = (const float*)d_in[10]; const float* ce_b1 = (const float*)d_in[11];
  const float* ce_w2 = (const float*)d_in[12]; const float* ce_b2 = (const float*)d_in[13];
  const float* cd_w1 = (const float*)d_in[14]; const float* cd_b1 = (const float*)d_in[15];
  const float* cd_w2 = (const float*)d_in[16]; const float* cd_b2 = (const float*)d_in[17];
  const float* be_w1 = (const float*)d_in[18]; const float* be_b1 = (const float*)d_in[19];
  const float* be_w2 = (const float*)d_in[20]; const float* be_b2 = (const float*)d_in[21];
  const float* bd_w1 = (const float*)d_in[22]; const float* bd_b1 = (const float*)d_in[23];
  const float* bd_w2 = (const float*)d_in[24]; const float* bd_b2 = (const float*)d_in[25];

  char* ws = (char*)d_ws;
  size_t off = 0;
  auto alloc = [&](size_t bytes) -> char* {
    char* p = ws + off;
    off = (off + bytes + 255) & ~(size_t)255;
    return p;
  };

  DecCtx c;
  c.stream = stream; c.f = f; c.r = r;
  c.w_llr_enc = w_llr_enc; c.b_llr_enc = b_llr_enc;
  c.w_llr_dec = w_llr_dec; c.b_llr_dec = b_llr_dec;
  c.ce_b1 = ce_b1; c.ce_b2 = ce_b2; c.cd_b1 = cd_b1; c.cd_b2 = cd_b2;
  c.be_b1 = be_b1; c.be_b2 = be_b2; c.bd_b1 = bd_b1; c.bd_b2 = bd_b2;
  c.out = (float*)d_out;

  for (int d = 0; d <= 9; ++d) {
    int nd = Nseq >> d;
    c.eco[d] = (__bf16*)alloc((size_t)Bsz * nd * Dm * sizeof(__bf16));
    c.ech[d] = (__bf16*)alloc((size_t)Bsz * nd * Dm * sizeof(__bf16));
    c.hpL[d] = (int*)alloc((size_t)Bsz * nd * sizeof(int));
    c.hpR[d] = (int*)alloc((size_t)Bsz * nd * sizeof(int));
  }
  __bf16* ceW1p = (__bf16*)alloc((size_t)2 * Dm * Hm * 2);
  __bf16* ceW2p = (__bf16*)alloc((size_t)Hm * Dm * 2);
  __bf16* cdW1p = (__bf16*)alloc((size_t)2 * Dm * Hm * 2);
  __bf16* cdW2p = (__bf16*)alloc((size_t)Hm * Dm * 2);
  __bf16* beW1p = (__bf16*)alloc((size_t)3 * Dm * Hm * 2);
  __bf16* beW2p = (__bf16*)alloc((size_t)Hm * Dm * 2);
  __bf16* bdW1p = (__bf16*)alloc((size_t)3 * Dm * Hm * 2);
  __bf16* bdW2p = (__bf16*)alloc((size_t)Hm * Dm * 2);
  __bf16* elab  = (__bf16*)alloc((size_t)2 * Dm * 2);
  c.ceW1p = ceW1p; c.ceW2p = ceW2p; c.cdW1p = cdW1p; c.cdW2p = cdW2p;
  c.beW1p = beW1p; c.beW2p = beW2p; c.bdW1p = bdW1p; c.bdW2p = bdW2p;
  c.elab = elab;

  auto pack = [&](const float* W, __bf16* Wp, int K, int Nc) {
    int total = K * Nc;
    pack_w_kernel<<<(total + 255) / 256, 256, 0, stream>>>(W, Wp, K, Nc);
  };
  pack(ce_w1, ceW1p, 2 * Dm, Hm); pack(ce_w2, ceW2p, Hm, Dm);
  pack(cd_w1, cdW1p, 2 * Dm, Hm); pack(cd_w2, cdW2p, Hm, Dm);
  pack(be_w1, beW1p, 3 * Dm, Hm); pack(be_w2, beW2p, Hm, Dm);
  pack(bd_w1, bdW1p, 3 * Dm, Hm); pack(bd_w2, bdW2p, Hm, Dm);
  pack_labels_kernel<<<1, 2 * Dm, 0, stream>>>(emb_labels, elab);

  {
    int total = Bsz * Nseq * (Dm / 8);
    init_emb_kernel<<<(total + 255) / 256, 256, 0, stream>>>(
        y, emb_obs_enc, emb_obs_dec, c.eco[0], c.ech[0]);
  }

  decode_rec(c, 0, 0, Nseq, c.hpL[0]);
}
